// AttentionPairBias_81604378624521
// MI455X (gfx1250) — matrix operand order repaired and checked by reading, not compiled, not run on hardware
//
#include <hip/hip_runtime.h>

typedef __attribute__((ext_vector_type(16))) _Float16 v16h;
typedef __attribute__((ext_vector_type(8)))  _Float16 v8h;
typedef __attribute__((ext_vector_type(8)))  float    v8f;

#define N_SEQ   768
#define C_DIM   384
#define CZ_DIM  128
#define H_HEADS 16
#define CH_DIM  24
#define CH_PAD  32

// ---------------------------------------------------------------- helpers
// Transposing convert: dst[n*K + k] = (f16)src[k*N + n]   (dst writes coalesced)
__global__ void f32_to_f16_t_kernel(const float* __restrict__ src,
                                    _Float16* __restrict__ dst, int N, int K) {
  int i = blockIdx.x * blockDim.x + threadIdx.x;
  if (i < N * K) {
    int n = i / K, k = i - n * K;
    dst[i] = (_Float16)src[(size_t)k * N + n];
  }
}

__global__ void gate_kernel(const float* __restrict__ g_raw,
                            const float* __restrict__ o,
                            _Float16* __restrict__ go, int n) {
  int i = blockIdx.x * blockDim.x + threadIdx.x;
  if (i < n) {
    float g = 1.0f / (1.0f + __expf(-g_raw[i]));
    go[i] = (_Float16)(g * o[i]);
  }
}

// ---------------------------------------------------------------- LN(a)
__global__ void ln_a_kernel(const float* __restrict__ a, const float* __restrict__ w,
                            const float* __restrict__ b, _Float16* __restrict__ out) {
  __shared__ float red[128];
  int row = blockIdx.x, tid = threadIdx.x;
  const float* x = a + (size_t)row * C_DIM;
  float s = 0.f;
  for (int c = tid; c < C_DIM; c += 128) s += x[c];
  red[tid] = s; __syncthreads();
  for (int off = 64; off > 0; off >>= 1) { if (tid < off) red[tid] += red[tid + off]; __syncthreads(); }
  float mu = red[0] * (1.0f / C_DIM);
  __syncthreads();
  float ss = 0.f;
  for (int c = tid; c < C_DIM; c += 128) { float d = x[c] - mu; ss += d * d; }
  red[tid] = ss; __syncthreads();
  for (int off = 64; off > 0; off >>= 1) { if (tid < off) red[tid] += red[tid + off]; __syncthreads(); }
  float inv = rsqrtf(red[0] * (1.0f / C_DIM) + 1e-5f);
  for (int c = tid; c < C_DIM; c += 128)
    out[(size_t)row * C_DIM + c] = (_Float16)((x[c] - mu) * inv * w[c] + b[c]);
}

// ---------------------------------------------------------------- generic WMMA GEMM
// B is stored TRANSPOSED: Bt[n*ldbt + k]. All fragment loads contiguous b128.
// mode 0: f32 store, D = acc*scale + bias[n]
// mode 1: f16 store into CH-padded [H][M][CH_PAD]  (q/k layout)
// mode 2: f16 store into CH-padded [H][CH_PAD][M]  (v transposed layout)
__global__ void wmma_gemm_kernel(const _Float16* __restrict__ A, const _Float16* __restrict__ Bt,
                                 void* __restrict__ dst, const float* __restrict__ bias,
                                 int M, int N, int K, int lda, int ldbt,
                                 int mode, float scale) {
  int lane = threadIdx.x & 31;
  int col  = lane & 15;     // N index / A row index
  int hi   = lane >> 4;     // lane group 0/1
  int tm = blockIdx.y * 16;
  int tn = blockIdx.x * 16;
  v8f acc = {};
  for (int k0 = 0; k0 < K; k0 += 32) {
    const v8h* ap = (const v8h*)(A + (size_t)(tm + col) * lda + k0 + hi * 16);
    v8h a0 = ap[0], a1 = ap[1];                       // halfs [0..8) and [8..16)
    v16h af;
#pragma unroll
    for (int h = 0; h < 8; ++h) { af[h] = a0[h]; af[h + 8] = a1[h]; }
    v16h bf = *(const v16h*)(Bt + (size_t)(tn + col) * ldbt + k0 + hi * 16);
    acc = __builtin_amdgcn_wmma_f32_16x16x32_f16(false, af, false, bf,
                                                 (short)0, acc, false, false);
  }
  if (mode == 0) {
    float* C = (float*)dst;
#pragma unroll
    for (int r = 0; r < 8; ++r) {
      int m = tm + r + hi * 8;
      int n = tn + col;
      C[(size_t)m * N + n] = acc[r] * scale + (bias ? bias[n] : 0.f);
    }
  } else if (mode == 1) {
    _Float16* P = (_Float16*)dst;
#pragma unroll
    for (int r = 0; r < 8; ++r) {
      int m = tm + r + hi * 8;
      int n = tn + col;
      int head = n / CH_DIM, ch = n - head * CH_DIM;
      P[((size_t)head * M + m) * CH_PAD + ch] = (_Float16)(acc[r] * scale);
    }
  } else {
    _Float16* P = (_Float16*)dst;
#pragma unroll
    for (int r = 0; r < 8; ++r) {
      int m = tm + r + hi * 8;
      int n = tn + col;
      int head = n / CH_DIM, ch = n - head * CH_DIM;
      P[((size_t)head * CH_PAD + ch) * M + m] = (_Float16)(acc[r] * scale);
    }
  }
}

// ---------------------------------------------------------------- fused LN(z) @ Wz
// One wave per 16 (i,j) pair-rows; single streaming pass over z (302 MB).
// Wzt is transposed: Wzt[h*CZ + c].  Bias output stored as f16 [H][N][N]
// (halves the 2nd-largest traffic source: 75 MB -> 38 MB round trip).
__global__ void pair_bias_kernel(const float* __restrict__ z, const float* __restrict__ lw,
                                 const float* __restrict__ lb, const _Float16* __restrict__ Wzt,
                                 _Float16* __restrict__ biasbuf) {
  __shared__ _Float16 tile[16][CZ_DIM];
  __shared__ float red[32][2];
  int lane = threadIdx.x & 31;
  int row  = lane & 15;
  int half = lane >> 4;
  size_t gp = (size_t)blockIdx.x * 16 + row;
  const float4* zr4 = (const float4*)(z + gp * CZ_DIM + half * 64);
  __builtin_prefetch(z + ((size_t)blockIdx.x * 16 + 16) * CZ_DIM, 0, 1);
  float s = 0.f, ss = 0.f;
  float xv[64];
#pragma unroll
  for (int t = 0; t < 16; ++t) {
    float4 v = zr4[t];
    xv[4 * t + 0] = v.x; xv[4 * t + 1] = v.y; xv[4 * t + 2] = v.z; xv[4 * t + 3] = v.w;
    s  += v.x + v.y + v.z + v.w;
    ss += v.x * v.x + v.y * v.y + v.z * v.z + v.w * v.w;
  }
  red[lane][0] = s; red[lane][1] = ss;
  __syncthreads();
  float S  = red[row][0] + red[row + 16][0];
  float SS = red[row][1] + red[row + 16][1];
  float mu  = S * (1.0f / CZ_DIM);
  float inv = rsqrtf(SS * (1.0f / CZ_DIM) - mu * mu + 1e-5f);
#pragma unroll
  for (int t = 0; t < 64; ++t) {
    int c = half * 64 + t;
    tile[row][c] = (_Float16)((xv[t] - mu) * inv * lw[c] + lb[c]);
  }
  __syncthreads();
  v8f acc = {};
#pragma unroll
  for (int k0 = 0; k0 < CZ_DIM; k0 += 32) {
    const v8h* tp = (const v8h*)(&tile[row][k0 + half * 16]);
    v8h a0 = tp[0], a1 = tp[1];
    v16h af;
#pragma unroll
    for (int h = 0; h < 8; ++h) { af[h] = a0[h]; af[h + 8] = a1[h]; }
    v16h bf = *(const v16h*)(Wzt + (size_t)row * CZ_DIM + k0 + half * 16);
    acc = __builtin_amdgcn_wmma_f32_16x16x32_f16(false, af, false, bf,
                                                 (short)0, acc, false, false);
  }
#pragma unroll
  for (int r = 0; r < 8; ++r) {
    int m = r + half * 8;
    size_t g2 = (size_t)blockIdx.x * 16 + m;
    int i = (int)(g2 / N_SEQ), j = (int)(g2 % N_SEQ);
    biasbuf[((size_t)row * N_SEQ + i) * N_SEQ + j] = (_Float16)acc[r];  // row == head
  }
}

// ---------------------------------------------------------------- attention core
// One wave per (head, 16-query tile): QK^T + bias -> LDS softmax -> PV.
// v16t layout: [H][CH_PAD][N_SEQ] so PV B-fragments are contiguous along keys.
__global__ void attention_kernel(const _Float16* __restrict__ q16, const _Float16* __restrict__ k16,
                                 const _Float16* __restrict__ v16t, const _Float16* __restrict__ biasbuf,
                                 const float* __restrict__ mask, float* __restrict__ o) {
  __shared__ float sc[16][N_SEQ];     // 48 KB; later reused in place for f16 probabilities
  __shared__ float red[32];
  __shared__ float invs[16];
  int lane = threadIdx.x & 31;
  int row  = lane & 15;
  int half = lane >> 4;
  int qt   = blockIdx.x;
  int head = blockIdx.y;
  const _Float16* Q  = q16  + (size_t)head * N_SEQ * CH_PAD;
  const _Float16* Km = k16  + (size_t)head * N_SEQ * CH_PAD;
  const _Float16* Vt = v16t + (size_t)head * CH_PAD * N_SEQ;
  // Q A-fragment is invariant over key tiles (K = CH padded to 32)
  const v8h* qp = (const v8h*)(Q + (size_t)(qt * 16 + row) * CH_PAD + half * 16);
  v8h q0 = qp[0], q1 = qp[1];
  v16h af;
#pragma unroll
  for (int h = 0; h < 8; ++h) { af[h] = q0[h]; af[h + 8] = q1[h]; }
  for (int kt = 0; kt < N_SEQ / 16; ++kt) {
    v16h bf = *(const v16h*)(Km + (size_t)(kt * 16 + row) * CH_PAD + half * 16);
    v8f acc = {};
    acc = __builtin_amdgcn_wmma_f32_16x16x32_f16(false, af, false, bf,
                                                 (short)0, acc, false, false);
#pragma unroll
    for (int r = 0; r < 8; ++r) {
      int m = r + half * 8;
      int key = kt * 16 + row;
      float mb = 1e9f * (mask[key] - 1.0f);
      float pb = (float)biasbuf[((size_t)head * N_SEQ + qt * 16 + m) * N_SEQ + key];
      sc[m][key] = acc[r] + pb + mb;
    }
  }
  __syncthreads();
  // softmax: lane pair (row, row+16) splits the 768-wide row
  int c0 = half * (N_SEQ / 2);
  float mx = -1e30f;
  for (int t = 0; t < N_SEQ / 2; ++t) mx = fmaxf(mx, sc[row][c0 + t]);
  red[lane] = mx; __syncthreads();
  mx = fmaxf(red[row], red[row + 16]);
  __syncthreads();
  float s = 0.f;
  for (int t = 0; t < N_SEQ / 2; ++t) {
    float e = __expf(sc[row][c0 + t] - mx);
    sc[row][c0 + t] = e; s += e;
  }
  red[lane] = s; __syncthreads();
  if (half == 0) invs[row] = 1.0f / (red[row] + red[row + 16]);
  __syncthreads();
  float pin = invs[row];
  // convert probability row to f16 IN PLACE (write float index t/2 <= read index t),
  // folding in the softmax normalization
  _Float16* prow = (_Float16*)&sc[row][0];
  for (int t = 0; t < N_SEQ / 2; ++t) {
    float e = sc[row][c0 + t];
    prow[c0 + t] = (_Float16)(e * pin);
  }
  __syncthreads();
  // PV: K = 768, two 16-wide channel blocks (covers CH=24, pad is zero)
  for (int cb = 0; cb < 2; ++cb) {
    v8f acc = {};
    for (int k0 = 0; k0 < N_SEQ; k0 += 32) {
      const v8h* pp = (const v8h*)(prow + k0 + half * 16);
      v8h p0 = pp[0], p1 = pp[1];
      v16h pa;
#pragma unroll
      for (int h = 0; h < 8; ++h) { pa[h] = p0[h]; pa[h + 8] = p1[h]; }
      v16h bf = *(const v16h*)(Vt + (size_t)(cb * 16 + row) * N_SEQ + k0 + half * 16);
      acc = __builtin_amdgcn_wmma_f32_16x16x32_f16(false, pa, false, bf,
                                                   (short)0, acc, false, false);
    }
#pragma unroll
    for (int r = 0; r < 8; ++r) {
      int m = qt * 16 + r + half * 8;
      int ch = cb * 16 + row;
      if (ch < CH_DIM)
        o[(size_t)m * (H_HEADS * CH_DIM) + head * CH_DIM + ch] = acc[r];
    }
  }
}

// ---------------------------------------------------------------- launch
extern "C" void kernel_launch(void* const* d_in, const int* in_sizes, int n_in,
                              void* d_out, int out_size, void* d_ws, size_t ws_size,
                              hipStream_t stream) {
  (void)in_sizes; (void)n_in; (void)out_size; (void)ws_size;
  const float* a     = (const float*)d_in[0];
  const float* z     = (const float*)d_in[1];
  const float* mask  = (const float*)d_in[2];
  const float* ln_aw = (const float*)d_in[3];
  const float* ln_ab = (const float*)d_in[4];
  const float* ln_zw = (const float*)d_in[5];
  const float* ln_zb = (const float*)d_in[6];
  const float* Wz    = (const float*)d_in[7];
  const float* Wq    = (const float*)d_in[8];
  const float* Wk    = (const float*)d_in[9];
  const float* Wv    = (const float*)d_in[10];
  const float* Wg    = (const float*)d_in[11];
  const float* bg    = (const float*)d_in[12];
  const float* Wo    = (const float*)d_in[13];
  const float* bo    = (const float*)d_in[14];
  float* out = (float*)d_out;

  char* ws = (char*)d_ws;                      // all offsets 256B-aligned
  _Float16* anf16 = (_Float16*)(ws + 0);       //  768*384*2
  _Float16* q16   = (_Float16*)(ws + 589824);  //  16*768*32*2
  _Float16* k16   = (_Float16*)(ws + 1376256);
  _Float16* v16t  = (_Float16*)(ws + 2162688); //  16*32*768*2  (transposed V)
  _Float16* Wq16t = (_Float16*)(ws + 2949120); //  384*384*2 transposed
  _Float16* Wk16t = (_Float16*)(ws + 3244032);
  _Float16* Wv16t = (_Float16*)(ws + 3538944);
  _Float16* Wg16t = (_Float16*)(ws + 3833856);
  _Float16* Wo16t = (_Float16*)(ws + 4128768);
  _Float16* Wz16t = (_Float16*)(ws + 4423680); //  16*128*2 transposed
  float*    g_raw = (float*)   (ws + 4427776); //  768*384*4
  float*    obuf  = (float*)   (ws + 5607424); //  768*384*4
  _Float16* go16  = (_Float16*)(ws + 6787072); //  768*384*2
  _Float16* biasbuf = (_Float16*)(ws + 7376896); // 16*768*768*2 -> end 26251264

  const int wn = C_DIM * C_DIM;
  f32_to_f16_t_kernel<<<(wn + 255) / 256, 256, 0, stream>>>(Wq, Wq16t, C_DIM, C_DIM);
  f32_to_f16_t_kernel<<<(wn + 255) / 256, 256, 0, stream>>>(Wk, Wk16t, C_DIM, C_DIM);
  f32_to_f16_t_kernel<<<(wn + 255) / 256, 256, 0, stream>>>(Wv, Wv16t, C_DIM, C_DIM);
  f32_to_f16_t_kernel<<<(wn + 255) / 256, 256, 0, stream>>>(Wg, Wg16t, C_DIM, C_DIM);
  f32_to_f16_t_kernel<<<(wn + 255) / 256, 256, 0, stream>>>(Wo, Wo16t, C_DIM, C_DIM);
  f32_to_f16_t_kernel<<<(H_HEADS * CZ_DIM + 255) / 256, 256, 0, stream>>>(Wz, Wz16t, H_HEADS, CZ_DIM);

  hipMemsetAsync(q16, 0, (size_t)3 * 786432, stream);   // zero CH pad of q/k/v

  ln_a_kernel<<<N_SEQ, 128, 0, stream>>>(a, ln_aw, ln_ab, anf16);

  dim3 gP(C_DIM / 16, N_SEQ / 16);
  const float qscale = 0.20412414523193154f;            // CH^-0.5
  wmma_gemm_kernel<<<gP, 32, 0, stream>>>(anf16, Wq16t, q16,   nullptr, N_SEQ, C_DIM, C_DIM, C_DIM, C_DIM, 1, qscale);
  wmma_gemm_kernel<<<gP, 32, 0, stream>>>(anf16, Wk16t, k16,   nullptr, N_SEQ, C_DIM, C_DIM, C_DIM, C_DIM, 1, 1.0f);
  wmma_gemm_kernel<<<gP, 32, 0, stream>>>(anf16, Wv16t, v16t,  nullptr, N_SEQ, C_DIM, C_DIM, C_DIM, C_DIM, 2, 1.0f);
  wmma_gemm_kernel<<<gP, 32, 0, stream>>>(anf16, Wg16t, g_raw, bg,      N_SEQ, C_DIM, C_DIM, C_DIM, C_DIM, 0, 1.0f);

  pair_bias_kernel<<<(N_SEQ * N_SEQ) / 16, 32, 0, stream>>>(z, ln_zw, ln_zb, Wz16t, biasbuf);

  dim3 gA(N_SEQ / 16, H_HEADS);
  attention_kernel<<<gA, 32, 0, stream>>>(q16, k16, v16t, biasbuf, mask, obuf);

  const int gn = N_SEQ * C_DIM;
  gate_kernel<<<(gn + 255) / 256, 256, 0, stream>>>(g_raw, obuf, go16, gn);

  wmma_gemm_kernel<<<gP, 32, 0, stream>>>(go16, Wo16t, out, bo, N_SEQ, C_DIM, C_DIM, C_DIM, C_DIM, 0, 1.0f);
}
